// footballNN_53919019434133
// MI455X (gfx1250) — matrix operand order repaired and checked by reading, not compiled, not run on hardware
//
#include <hip/hip_runtime.h>

typedef __attribute__((ext_vector_type(16))) _Float16 v16h;
typedef __attribute__((ext_vector_type(8)))  _Float16 v8h;
typedef __attribute__((ext_vector_type(8)))  float    v8f;

// ---------------------------------------------------------------------------
// Problem constants
// ---------------------------------------------------------------------------
#define B_SZ     2048
#define T_SZ     128
#define F_WEEK   256
#define IN_R     960
#define HID      80
#define G4       320            // 4*HID gate width
#define HID_PAD  96             // HID padded to K multiple of 32

// fragment counts (Ntiles x Ktiles), 1 KB per fragment (32 lanes * 32B)
#define W1_KT   30   // 960/32
#define W1_NT   60   // 960/16
#define W2_KT   30
#define W2_NT   5    // 80/16
#define WIH_KT  8    // 256/32
#define WIH_NT  20   // 320/16
#define WHH_KT  3    // 96/32
#define WHH_NT  20

#define FRAG_BYTES 1024
// ws layout (bytes)
#define OFF_W1   0
#define OFF_W2   (OFF_W1  + (size_t)W1_NT  * W1_KT  * FRAG_BYTES)   // 1800 KB
#define OFF_WIH  (OFF_W2  + (size_t)W2_NT  * W2_KT  * FRAG_BYTES)   // +150 KB
#define OFF_WHH  (OFF_WIH + (size_t)WIH_NT * WIH_KT * FRAG_BYTES)   // +160 KB
#define OFF_R    (OFF_WHH + (size_t)WHH_NT * WHH_KT * FRAG_BYTES)   // +60 KB

// ---------------------------------------------------------------------------
// Math helpers — v_rcp_f32 (TRANS, co-executes with WMMA pipe) instead of the
// correctly-rounded v_div_scale/v_div_fixup sequence.
// ---------------------------------------------------------------------------
__device__ __forceinline__ float sigm(float x) {
    return __builtin_amdgcn_rcpf(1.0f + __expf(-x));
}
__device__ __forceinline__ float tanh_fast(float x) {
    return 2.0f * __builtin_amdgcn_rcpf(1.0f + __expf(-2.0f * x)) - 1.0f;
}

// Gather an A-fragment (16x32, f16) from an LDS panel laid out row-major
// [16][strideH] halves.  WMMA 16-bit A layout:
//   lanes 0-15 : row M=lane,    K in {k0+0..7, k0+16..23}
//   lanes 16-31: row M=lane-16, K in {k0+8..15, k0+24..31}
__device__ __forceinline__ v16h lds_a_frag(const _Float16* base, int strideH,
                                           int kstep, int lane) {
    int row = lane & 15;
    int off = row * strideH + kstep * 32 + ((lane >> 4) << 3);
    v8h lo = *(const v8h*)(base + off);
    v8h hi = *(const v8h*)(base + off + 16);
    return __builtin_shufflevector(lo, hi, 0,1,2,3,4,5,6,7,8,9,10,11,12,13,14,15);
}

#define WMMA_F16(A, Bv, Cv) \
    __builtin_amdgcn_wmma_f32_16x16x32_f16(false, (A), false, (Bv), (short)0, (Cv), false, false)

// ---------------------------------------------------------------------------
// Pack kernel: f32 weight [Ntot, Ksrc] row-major  ->  f16 WMMA B-fragments.
// B element (k, n) = W[n, k].  B 16-bit layout: lane L holds N = L&15,
// K = (e<8 ? e : e+8) + (L>=16 ? 8 : 0) for e in 0..15.  One wave per fragment.
// ---------------------------------------------------------------------------
__global__ __launch_bounds__(256)
void pack_bfrag_kernel(const float* __restrict__ W, _Float16* __restrict__ dst,
                       int Ktiles, int Ksrc, int nFrags) {
    int wv = threadIdx.x >> 5, lane = threadIdx.x & 31;
    int f = blockIdx.x * 8 + wv;
    if (f >= nFrags) return;
    int nT = f / Ktiles, kS = f % Ktiles;
    int row   = nT * 16 + (lane & 15);
    int kbase = kS * 32 + ((lane >> 4) << 3);
    v16h v;
#pragma unroll
    for (int e = 0; e < 16; ++e) {
        int k = kbase + (e < 8 ? e : e + 8);
        float x = (k < Ksrc) ? W[(size_t)row * Ksrc + k] : 0.0f;
        v[e] = (_Float16)x;
    }
    *((v16h*)dst + (size_t)f * 32 + lane) = v;
}

// ---------------------------------------------------------------------------
// MLP kernel: r = sig(sig(X W1^T + b1) W2^T + b2);  out0 = r W_fc^T + b_fc
// One block = 16 batch rows, 8 waves.
// ---------------------------------------------------------------------------
__global__ __launch_bounds__(256)
void mlp_kernel(const float* __restrict__ timeless,
                const _Float16* __restrict__ fragW1,
                const _Float16* __restrict__ fragW2,
                const float* __restrict__ b1, const float* __restrict__ b2,
                const float* __restrict__ Wfc, const float* __restrict__ bfc,
                float* __restrict__ rOut, float* __restrict__ out) {
    __shared__ _Float16 lds_a [16 * IN_R];   // 30 KB, reused as f32 r later
    __shared__ _Float16 lds_r1[16 * IN_R];   // 30 KB
    const int b0 = blockIdx.x * 16;
    const int tid = threadIdx.x, lane = tid & 31, wv = tid >> 5;
    const int col = lane & 15, hi = lane >> 4;

    // stage X tile [16,960] -> f16 LDS
    for (int i = tid; i < 16 * IN_R; i += 256) {
        int r = i / IN_R, c = i % IN_R;
        lds_a[i] = (_Float16)timeless[(size_t)(b0 + r) * IN_R + c];
    }
    __syncthreads();

    // layer 1: 60 N-tiles, K = 30 steps
    for (int n = wv; n < W1_NT; n += 8) {
        v8f acc = {};
#pragma unroll 6
        for (int k = 0; k < W1_KT; ++k) {
            v16h a = lds_a_frag(lds_a, IN_R, k, lane);
            v16h b = *((const v16h*)fragW1 + (size_t)(n * W1_KT + k) * 32 + lane);
            acc = WMMA_F16(a, b, acc);
        }
        float bias = b1[n * 16 + col];
#pragma unroll
        for (int r = 0; r < 8; ++r) {
            int row = r + (hi << 3);
            lds_r1[row * IN_R + n * 16 + col] = (_Float16)sigm(acc[r] + bias);
        }
    }
    __syncthreads();

    // layer 2: 5 N-tiles (waves 0-4), K = 30 steps over r1
    float* rloc = (float*)lds_a;             // reuse region: [16][80] f32
    if (wv < W2_NT) {
        int n = wv;
        v8f acc = {};
#pragma unroll 6
        for (int k = 0; k < W2_KT; ++k) {
            v16h a = lds_a_frag(lds_r1, IN_R, k, lane);
            v16h b = *((const v16h*)fragW2 + (size_t)(n * W2_KT + k) * 32 + lane);
            acc = WMMA_F16(a, b, acc);
        }
        float bias = b2[n * 16 + col];
#pragma unroll
        for (int r = 0; r < 8; ++r) {
            int row = r + (hi << 3);
            float s = sigm(acc[r] + bias);
            rOut[(size_t)(b0 + row) * HID + n * 16 + col] = s;
            rloc[row * HID + n * 16 + col] = s;
        }
    }
    __syncthreads();

    // out0 = r W_fc^T + b_fc  (t = 0 slice)
    if (tid < 32) {
        int row = tid & 15, o = tid >> 4;
        float acc = bfc[o];
        for (int j = 0; j < HID; ++j) acc += rloc[row * HID + j] * Wfc[o * HID + j];
        out[(size_t)(b0 + row) * 2 + o] = acc;
    }
}

// ---------------------------------------------------------------------------
// LSTM kernel: per-block 16 batch rows for all 128 steps (recurrence is
// independent per row).  10 waves; wave w owns gate N-tiles 2w and 2w+1.
// W_hh fragments (time-invariant, 48 VGPRs) are pinned in registers.
// ---------------------------------------------------------------------------
__global__ __launch_bounds__(320)
void lstm_kernel(const float* __restrict__ X,        // [B, T, F_WEEK]
                 const _Float16* __restrict__ fragWih,
                 const _Float16* __restrict__ fragWhh,
                 const float* __restrict__ b_ih, const float* __restrict__ b_hh,
                 const float* __restrict__ Wfc,  const float* __restrict__ bfc,
                 const float* __restrict__ rIn,  // [B, HID] initial h = c
                 float* __restrict__ out) {
    __shared__ _Float16 lds_x[16 * F_WEEK];   //  8 KB
    __shared__ _Float16 lds_h[16 * HID_PAD];  //  3 KB (cols 80..95 zero)
    __shared__ float    lds_c[16 * HID];      //  5 KB
    __shared__ float    lds_g[16 * G4];       // 20 KB
    __shared__ float    lds_o[16 * 2];

    const int b0 = blockIdx.x * 16;
    const int tid = threadIdx.x, lane = tid & 31, wv = tid >> 5;
    const int col = lane & 15, hi = lane >> 4;
    const int n0 = 2 * wv, n1 = 2 * wv + 1;
    const float bias0 = b_ih[n0 * 16 + col] + b_hh[n0 * 16 + col];
    const float bias1 = b_ih[n1 * 16 + col] + b_hh[n1 * 16 + col];

    // pin the time-invariant W_hh fragments in VGPRs (2 tiles x 3 K-steps)
    v16h whh0[WHH_KT], whh1[WHH_KT];
#pragma unroll
    for (int k = 0; k < WHH_KT; ++k) {
        whh0[k] = *((const v16h*)fragWhh + (size_t)(n0 * WHH_KT + k) * 32 + lane);
        whh1[k] = *((const v16h*)fragWhh + (size_t)(n1 * WHH_KT + k) * 32 + lane);
    }

    // init h = c = r
    for (int i = tid; i < 16 * HID; i += 320) {
        int r = i / HID, j = i % HID;
        float v = rIn[(size_t)(b0 + r) * HID + j];
        lds_c[i] = v;
        lds_h[r * HID_PAD + j] = (_Float16)v;
    }
    for (int i = tid; i < 16 * 16; i += 320)   // zero K-pad 80..95
        lds_h[(i / 16) * HID_PAD + HID + (i % 16)] = (_Float16)0.0f;
    __syncthreads();

    for (int t = 0; t < T_SZ; ++t) {
        // stage x_t tile [16,256] -> f16 LDS (float4 granules); init head accum
        for (int i = tid; i < 16 * F_WEEK / 4; i += 320) {
            int r = i >> 6, c4 = (i & 63) << 2;
            const float4 v = *(const float4*)(X + ((size_t)(b0 + r) * T_SZ + t) * F_WEEK + c4);
            _Float16* d = lds_x + r * F_WEEK + c4;
            d[0] = (_Float16)v.x; d[1] = (_Float16)v.y;
            d[2] = (_Float16)v.z; d[3] = (_Float16)v.w;
        }
        if (tid < 32) lds_o[tid] = bfc[tid >> 4];
        if (t + 1 < T_SZ && tid < 16)   // warm L2/WGP$ for next step
            __builtin_prefetch(X + ((size_t)(b0 + tid) * T_SZ + t + 1) * F_WEEK, 0, 3);
        __syncthreads();

        // gates[16,320] = x W_ih^T + h W_hh^T + (b_ih + b_hh)
        v8f a0, a1;
#pragma unroll
        for (int r = 0; r < 8; ++r) { a0[r] = bias0; a1[r] = bias1; }
#pragma unroll
        for (int k = 0; k < WIH_KT; ++k) {
            v16h a  = lds_a_frag(lds_x, F_WEEK, k, lane);
            v16h bA = *((const v16h*)fragWih + (size_t)(n0 * WIH_KT + k) * 32 + lane);
            v16h bB = *((const v16h*)fragWih + (size_t)(n1 * WIH_KT + k) * 32 + lane);
            a0 = WMMA_F16(a, bA, a0);
            a1 = WMMA_F16(a, bB, a1);
        }
#pragma unroll
        for (int k = 0; k < WHH_KT; ++k) {
            v16h a = lds_a_frag(lds_h, HID_PAD, k, lane);
            a0 = WMMA_F16(a, whh0[k], a0);
            a1 = WMMA_F16(a, whh1[k], a1);
        }
#pragma unroll
        for (int r = 0; r < 8; ++r) {
            int row = r + (hi << 3);
            lds_g[row * G4 + n0 * 16 + col] = a0[r];
            lds_g[row * G4 + n1 * 16 + col] = a1[r];
        }
        __syncthreads();

        // pointwise LSTM update + 2-wide head (LDS-atomic reduction)
        for (int i = tid; i < 16 * HID; i += 320) {
            int row = i / HID, j = i % HID;
            const float* g = lds_g + row * G4;
            float iv = sigm(g[j]);
            float fv = sigm(g[HID + j]);
            float gv = tanh_fast(g[2 * HID + j]);
            float ov = sigm(g[3 * HID + j]);
            float c = fv * lds_c[i] + iv * gv;
            lds_c[i] = c;
            float h = ov * tanh_fast(c);
            lds_h[row * HID_PAD + j] = (_Float16)h;
            atomicAdd(&lds_o[row * 2 + 0], h * Wfc[j]);
            atomicAdd(&lds_o[row * 2 + 1], h * Wfc[HID + j]);
        }
        __syncthreads();

        if (tid < 32) {
            int row = tid & 15, o = tid >> 4;
            out[((size_t)(t + 1) * B_SZ + b0 + row) * 2 + o] = lds_o[tid];
        }
        __syncthreads();
    }
}

// ---------------------------------------------------------------------------
// Launch
// ---------------------------------------------------------------------------
extern "C" void kernel_launch(void* const* d_in, const int* in_sizes, int n_in,
                              void* d_out, int out_size, void* d_ws, size_t ws_size,
                              hipStream_t stream) {
    (void)in_sizes; (void)n_in; (void)out_size; (void)ws_size;
    const float* timeless = (const float*)d_in[0];
    const float* game     = (const float*)d_in[1];
    const float* W1   = (const float*)d_in[2];
    const float* b1   = (const float*)d_in[3];
    const float* W2   = (const float*)d_in[4];
    const float* b2   = (const float*)d_in[5];
    const float* W_ih = (const float*)d_in[6];
    const float* b_ih = (const float*)d_in[7];
    const float* W_hh = (const float*)d_in[8];
    const float* b_hh = (const float*)d_in[9];
    const float* W_fc = (const float*)d_in[10];
    const float* b_fc = (const float*)d_in[11];
    float* out = (float*)d_out;

    char* ws = (char*)d_ws;
    _Float16* fragW1  = (_Float16*)(ws + OFF_W1);
    _Float16* fragW2  = (_Float16*)(ws + OFF_W2);
    _Float16* fragWih = (_Float16*)(ws + OFF_WIH);
    _Float16* fragWhh = (_Float16*)(ws + OFF_WHH);
    float*    rPanel  = (float*)   (ws + OFF_R);

    // weight pre-convert + fragment swizzle (stays resident in L2)
    pack_bfrag_kernel<<<(W1_NT * W1_KT + 7) / 8, 256, 0, stream>>>(W1,  fragW1,  W1_KT,  IN_R,  W1_NT * W1_KT);
    pack_bfrag_kernel<<<(W2_NT * W2_KT + 7) / 8, 256, 0, stream>>>(W2,  fragW2,  W2_KT,  IN_R,  W2_NT * W2_KT);
    pack_bfrag_kernel<<<(WIH_NT * WIH_KT + 7) / 8, 256, 0, stream>>>(W_ih, fragWih, WIH_KT, F_WEEK, WIH_NT * WIH_KT);
    pack_bfrag_kernel<<<(WHH_NT * WHH_KT + 7) / 8, 256, 0, stream>>>(W_hh, fragWhh, WHH_KT, HID,   WHH_NT * WHH_KT);

    mlp_kernel<<<B_SZ / 16, 256, 0, stream>>>(timeless, fragW1, fragW2, b1, b2,
                                              W_fc, b_fc, rPanel, out);
    lstm_kernel<<<B_SZ / 16, 320, 0, stream>>>(game, fragWih, fragWhh, b_ih, b_hh,
                                               W_fc, b_fc, rPanel, out);
}
